// MulModel_GCN_GCN_RGCN_LLM_65249143161441
// MI455X (gfx1250) — compile-verified, matrix operand order checked by reading, not run on hardware
//
#include <hip/hip_runtime.h>
#include <hip/hip_bf16.h>

// ---------------- model constants ----------------
#define BATCH 8
#define NPG   4096
#define DIM   768
#define EPG   32768
#define NEDGE (BATCH*EPG)          // 262144
#define KP1   3277                 // ceil(0.8*4096)
#define KP2   2622                 // ceil(0.8*3277)
#define N1    (BATCH*NPG)          // 32768
#define N2    (BATCH*KP1)          // 26216
#define N3    (BATCH*KP2)          // 20976
#define LRELU 0.01f
#define CDIV(a,b) (((a)+(b)-1)/(b))

typedef __attribute__((ext_vector_type(16))) __bf16 v16bf;
typedef __attribute__((ext_vector_type(8)))  __bf16 v8bf;
typedef __attribute__((ext_vector_type(8)))  float  v8f;

__device__ __forceinline__ float leaky(float v) { return v > 0.f ? v : LRELU * v; }

// ---------------- utility kernels ----------------
__global__ void fill_f32(float* p, float v, long n) {
  long i = blockIdx.x * (long)blockDim.x + threadIdx.x;
  if (i < n) p[i] = v;
}

__global__ void init_edges(const int* __restrict__ ei, int* src, int* dst, float* em, int ne) {
  int e = blockIdx.x * blockDim.x + threadIdx.x;
  if (e < ne) { src[e] = ei[e]; dst[e] = ei[ne + e]; em[e] = 1.f; }
}

// Convert f32 weights -> bf16 in WMMA-B fragment order:
// tile (kt = k/32, t = c/16) is 512 contiguous bf16: [half*16 + n][slot],
// slot = (k&7) + 8*((k>>4)&1), half = (k>>3)&1.  One GEMM B-stage then is a
// straight contiguous global->LDS copy and each lane fragment one 32B read.
__global__ void permute_w_bf16(const float* __restrict__ W, __bf16* __restrict__ out,
                               int K, int N) {
  long t = blockIdx.x * (long)blockDim.x + threadIdx.x;
  if (t >= (long)K * N) return;
  int k = (int)(t / N), c = (int)(t % N);
  int kt = k >> 5, kk = k & 31;
  int half = (kk >> 3) & 1;
  int slot = (kk & 7) + ((kk >> 4) & 1) * 8;
  int tl = c >> 4, n = c & 15;
  out[((size_t)(kt * (N >> 4) + tl)) * 512 + ((half * 16 + n) << 4) + slot] = (__bf16)W[t];
}

// ---------------- WMMA bf16 GEMM: C[M,N] = A[M,K] * Wperm[K,N] ----------------
// 256 threads = 8 waves; block tile 256x64; K step 32.  Each wave owns a
// 32-row stripe: 2 A fragments x 4 B tiles -> 8 v_wmma_f32_16x16x32_bf16 per
// K-step, with all fragments loaded as contiguous 32B LDS reads.
__global__ __launch_bounds__(256)
void gemm_wmma_bf16(const float* __restrict__ A, const __bf16* __restrict__ Wp,
                    float* __restrict__ C, int M, int K, int N)
{
  __shared__ alignas(32) __bf16 sA[256 * 32];  // fragment order: [sub16][laneRow32][16]
  __shared__ alignas(32) __bf16 sB[4 * 512];   // 4 n-tiles, fragment order
  const int tid  = threadIdx.x;
  const int wave = tid >> 5;
  const int lane = tid & 31;
  const int bm   = blockIdx.x * 256;
  const int bn   = blockIdx.y * 64;
  const int ntile0  = bn >> 4;
  const int nTilesN = N >> 4;

  v8f acc[2][4] = {};
  for (int k0 = 0; k0 < K; k0 += 32) {
    const int kt = k0 >> 5;
    // ---- stage A: 256x32 f32 -> bf16, fragment order. 4 chunks of 8 / thread.
    #pragma unroll
    for (int i = 0; i < 4; ++i) {
      int chunk = tid + i * 256;              // 0..1023
      int r = chunk >> 2;                     // tile row 0..255
      int q = chunk & 3;                      // which 8-wide k chunk
      int gr = bm + r;
      float4 f0 = {0.f,0.f,0.f,0.f}, f1 = {0.f,0.f,0.f,0.f};
      if (gr < M) {
        const float4* g = (const float4*)(A + (size_t)gr * K + k0 + q * 8);
        f0 = g[0]; f1 = g[1];
      }
      v8bf pk;
      pk[0]=(__bf16)f0.x; pk[1]=(__bf16)f0.y; pk[2]=(__bf16)f0.z; pk[3]=(__bf16)f0.w;
      pk[4]=(__bf16)f1.x; pk[5]=(__bf16)f1.y; pk[6]=(__bf16)f1.z; pk[7]=(__bf16)f1.w;
      int sub = r >> 4, m = r & 15;
      int half = q & 1;                       // q0:k0..7(h0) q1:k8..15(h1) q2:k16..23(h0) q3:k24..31(h1)
      int slotbase = (q & 2) * 4;             // 0 or 8
      *(v8bf*)&sA[((sub * 32 + half * 16 + m) << 4) + slotbase] = pk;
    }
    // ---- stage B: 4 contiguous 1KB tile copies (pre-permuted in global)
    {
      int t  = tid >> 6;
      int o8 = (tid & 63) << 3;
      const v8bf* src = (const v8bf*)(Wp + ((size_t)(kt * nTilesN + ntile0 + t)) * 512 + o8);
      *(v8bf*)&sB[t * 512 + o8] = *src;
    }
    __syncthreads();
    // ---- fragments: one 32B contiguous LDS read each
    v16bf a0 = *(const v16bf*)&sA[((wave * 2 + 0) * 32 + lane) << 4];
    v16bf a1 = *(const v16bf*)&sA[((wave * 2 + 1) * 32 + lane) << 4];
    #pragma unroll
    for (int t = 0; t < 4; ++t) {
      v16bf bfrag = *(const v16bf*)&sB[(t * 32 + lane) << 4];
      acc[0][t] = __builtin_amdgcn_wmma_f32_16x16x32_bf16(
          false, a0, false, bfrag, (short)0, acc[0][t], false, false);
      acc[1][t] = __builtin_amdgcn_wmma_f32_16x16x32_bf16(
          false, a1, false, bfrag, (short)0, acc[1][t], false, false);
    }
    __syncthreads();
  }
  // C/D layout: VGPR i -> M = i + 8*(lane/16), N = lane%16
  const int half = lane >> 4, nn = lane & 15;
  #pragma unroll
  for (int s = 0; s < 2; ++s)
    #pragma unroll
    for (int t = 0; t < 4; ++t)
      #pragma unroll
      for (int i = 0; i < 8; ++i) {
        int gr = bm + (wave * 2 + s) * 16 + i + 8 * half;
        int gc = bn + t * 16 + nn;
        if (gr < M) C[(size_t)gr * N + gc] = acc[s][t][i];
      }
}

// ---------------- GCN aggregation (bandwidth bound) ----------------
__global__ void scatter_deg(const int* __restrict__ dst, const float* __restrict__ em,
                            float* deg, int ne) {
  int e = blockIdx.x * blockDim.x + threadIdx.x;
  if (e < ne && em[e] != 0.f) atomicAdd(deg + dst[e], em[e]);
}

__global__ void rsqrt_k(const float* __restrict__ deg, float* dis, int n) {
  int i = blockIdx.x * blockDim.x + threadIdx.x;
  if (i < n) dis[i] = rsqrtf(deg[i]);
}

__global__ void gcn_scatter(const float4* __restrict__ H4, const int* __restrict__ src,
                            const int* __restrict__ dst, const float* __restrict__ em,
                            const float* __restrict__ dis, float* AG, int ne, int c4) {
  long t = blockIdx.x * (long)blockDim.x + threadIdx.x;
  if (t >= (long)ne * c4) return;
  int e = (int)(t / c4), c = (int)(t % c4);
  float m = em[e];
  if (m == 0.f) return;
  int s = src[e], d = dst[e];
  float coef = dis[s] * dis[d] * m;
  float4 h = H4[(size_t)s * c4 + c];
  float* o = AG + ((size_t)d * c4 + c) * 4;
  atomicAdd(o + 0, h.x * coef); atomicAdd(o + 1, h.y * coef);
  atomicAdd(o + 2, h.z * coef); atomicAdd(o + 3, h.w * coef);
}

__global__ void gcn_finalize(const float* __restrict__ AG, const float* __restrict__ H,
                             const float* __restrict__ dis, const float* __restrict__ bias,
                             float* X, int n, int Cf) {
  long t = blockIdx.x * (long)blockDim.x + threadIdx.x;
  if (t >= (long)n * Cf) return;
  int i = (int)(t / Cf), c = (int)(t % Cf);
  float d2 = dis[i] * dis[i];
  X[t] = leaky(AG[t] + H[t] * d2 + bias[c]);
}

// ---------------- RGCN aggregation ----------------
__global__ void rgcn_scatter(const float4* __restrict__ H4, const int* __restrict__ src,
                             const int* __restrict__ dst, const float* __restrict__ em,
                             const int* __restrict__ et, int rel,
                             float* S, int ne, int c4) {
  long t = blockIdx.x * (long)blockDim.x + threadIdx.x;
  if (t >= (long)ne * c4) return;
  int e = (int)(t / c4), c = (int)(t % c4);
  float m = (et[e] == rel) ? em[e] : 0.f;
  if (m == 0.f) return;
  int s = src[e], d = dst[e];
  float4 h = H4[(size_t)s * c4 + c];
  float* o = S + ((size_t)d * c4 + c) * 4;
  atomicAdd(o + 0, h.x * m); atomicAdd(o + 1, h.y * m);
  atomicAdd(o + 2, h.z * m); atomicAdd(o + 3, h.w * m);
}

__global__ void rgcn_cnt(const int* __restrict__ dst, const float* __restrict__ em,
                         const int* __restrict__ et, int rel, float* cnt, int ne) {
  int e = blockIdx.x * blockDim.x + threadIdx.x;
  if (e < ne) {
    float m = (et[e] == rel) ? em[e] : 0.f;
    if (m != 0.f) atomicAdd(cnt + dst[e], m);
  }
}

__global__ void rgcn_combine(float* AG, const float* __restrict__ S,
                             const float* __restrict__ cnt, int n, int Cf) {
  long t = blockIdx.x * (long)blockDim.x + threadIdx.x;
  if (t >= (long)n * Cf) return;
  int i = (int)(t / Cf);
  AG[t] += S[t] / fmaxf(cnt[i], 1.f);
}

__global__ void rgcn_finalize(const float* __restrict__ AG, const float* __restrict__ bias,
                              float* X, int n, int Cf) {
  long t = blockIdx.x * (long)blockDim.x + threadIdx.x;
  if (t >= (long)n * Cf) return;
  int c = (int)(t % Cf);
  X[t] = leaky(AG[t] + bias[c]);
}

// ---------------- SAGPool ----------------
__global__ void gemv_k(const float* __restrict__ X, const float* __restrict__ w,
                       const float* __restrict__ bias, float* out, int n, int Cf) {
  int i = blockIdx.x * blockDim.x + threadIdx.x;
  if (i >= n) return;
  float s = bias ? bias[0] : 0.f;
  for (int c = 0; c < Cf; ++c) s += X[(size_t)i * Cf + c] * w[c];
  out[i] = s;
}

__global__ void scatter_scalar(const float* __restrict__ t, const int* __restrict__ src,
                               const int* __restrict__ dst, const float* __restrict__ em,
                               float* out, int ne) {
  int e = blockIdx.x * blockDim.x + threadIdx.x;
  if (e < ne && em[e] != 0.f) atomicAdd(out + dst[e], t[src[e]] * em[e]);
}

__global__ void score_k(const float* __restrict__ sagg, const float* __restrict__ troot,
                        const float* __restrict__ relb, float* score, int n) {
  int i = blockIdx.x * blockDim.x + threadIdx.x;
  if (i < n) score[i] = tanhf(sagg[i] + relb[0] + troot[i]);
}

// per-graph bitonic top-k (ng <= 4096); one block per graph
__global__ __launch_bounds__(1024)
void topk_sort(const float* __restrict__ score, int ng, int k, int* perm, int* newidx) {
  __shared__ float ss[4096];
  __shared__ int   si[4096];
  const int b = blockIdx.x;
  for (int i = threadIdx.x; i < 4096; i += 1024) {
    ss[i] = (i < ng) ? score[b * ng + i] : -1e30f;
    si[i] = i;
  }
  __syncthreads();
  for (int ksz = 2; ksz <= 4096; ksz <<= 1) {
    for (int j = ksz >> 1; j > 0; j >>= 1) {
      for (int i = threadIdx.x; i < 4096; i += 1024) {
        int ixj = i ^ j;
        if (ixj > i) {
          bool up = ((i & ksz) == 0);           // descending overall
          bool sw = up ? (ss[i] < ss[ixj]) : (ss[i] > ss[ixj]);
          if (sw) {
            float tf = ss[i]; ss[i] = ss[ixj]; ss[ixj] = tf;
            int   ti = si[i]; si[i] = si[ixj]; si[ixj] = ti;
          }
        }
      }
      __syncthreads();
    }
  }
  for (int i = threadIdx.x; i < ng; i += 1024) newidx[b * ng + i] = -1;
  __syncthreads();
  for (int i = threadIdx.x; i < k; i += 1024) {
    int node = b * ng + si[i];
    perm[b * k + i] = node;
    newidx[node] = b * k + i;
  }
}

__global__ void pool_gather(const float* __restrict__ Xin, const float* __restrict__ score,
                            const int* __restrict__ perm, float* Xout, int nk, int Cf) {
  long t = blockIdx.x * (long)blockDim.x + threadIdx.x;
  if (t >= (long)nk * Cf) return;
  int j = (int)(t / Cf), c = (int)(t % Cf);
  int p = perm[j];
  Xout[t] = Xin[(size_t)p * Cf + c] * score[p];
}

__global__ void remap_edges(int* src, int* dst, float* em, const int* __restrict__ newidx, int ne) {
  int e = blockIdx.x * blockDim.x + threadIdx.x;
  if (e >= ne) return;
  int s = newidx[src[e]], d = newidx[dst[e]];
  float keep = (s >= 0 && d >= 0) ? em[e] : 0.f;
  src[e] = s < 0 ? 0 : s;
  dst[e] = d < 0 ? 0 : d;
  em[e]  = keep;
}

// ---------------- attention readout ----------------
__global__ __launch_bounds__(256)
void softmax_pg(float* g, int k) {
  __shared__ float red[256];
  const int b = blockIdx.x;
  float mx = -1e30f;
  for (int i = threadIdx.x; i < k; i += 256) mx = fmaxf(mx, g[b * k + i]);
  red[threadIdx.x] = mx; __syncthreads();
  for (int s = 128; s > 0; s >>= 1) {
    if (threadIdx.x < s) red[threadIdx.x] = fmaxf(red[threadIdx.x], red[threadIdx.x + s]);
    __syncthreads();
  }
  mx = red[0]; __syncthreads();
  float sum = 0.f;
  for (int i = threadIdx.x; i < k; i += 256) {
    float e = __expf(g[b * k + i] - mx);
    g[b * k + i] = e; sum += e;
  }
  red[threadIdx.x] = sum; __syncthreads();
  for (int s = 128; s > 0; s >>= 1) {
    if (threadIdx.x < s) red[threadIdx.x] += red[threadIdx.x + s];
    __syncthreads();
  }
  sum = red[0]; __syncthreads();
  for (int i = threadIdx.x; i < k; i += 256) g[b * k + i] /= sum;
}

__global__ void att_wsum(const float* __restrict__ X, const float* __restrict__ w,
                         float* R, int k, int Cf, int colOff) {
  int c = blockIdx.x * blockDim.x + threadIdx.x;
  int b = blockIdx.y;
  if (c >= Cf) return;
  float s = 0.f;
  for (int i = 0; i < k; ++i) s += w[b * k + i] * X[((size_t)(b * k + i)) * Cf + c];
  R[b * 1024 + colOff + c] = s;
}

// ---------------- MLP head ----------------
__global__ void small_gemm(const float* __restrict__ A, const float* __restrict__ W,
                           const float* __restrict__ bias, float* Cm,
                           int M, int K, int N, int ldc, int colOff) {
  int t = blockIdx.x * blockDim.x + threadIdx.x;
  if (t >= M * N) return;
  int i = t / N, j = t % N;
  float s = bias ? bias[j] : 0.f;
  for (int q = 0; q < K; ++q) s += A[i * K + q] * W[q * N + j];
  Cm[i * ldc + colOff + j] = s;
}

__global__ void bn_leaky(float* Z, const float* __restrict__ g, const float* __restrict__ bb,
                         int M, int Ncol) {
  int j = blockIdx.x * blockDim.x + threadIdx.x;
  if (j >= Ncol) return;
  float mu = 0.f;
  for (int i = 0; i < M; ++i) mu += Z[i * Ncol + j];
  mu /= M;
  float var = 0.f;
  for (int i = 0; i < M; ++i) { float d = Z[i * Ncol + j] - mu; var += d * d; }
  var /= M;
  float inv = rsqrtf(var + 1e-5f);
  for (int i = 0; i < M; ++i) {
    float y = (Z[i * Ncol + j] - mu) * inv * g[j] + bb[j];
    Z[i * Ncol + j] = leaky(y);
  }
}

// ---------------- host-side orchestration ----------------
struct Bufs {
  float *XA, *XB, *HB, *AG, *SB;
  __bf16* WBF;
  float *deg, *dis, *cnt, *tn, *tn2, *sagg, *score, *gate;
  int *perm, *newidx, *esrc, *edst;
  float *emask, *R;
};

static void gemm(hipStream_t st, const float* A, const float* W, __bf16* Wbf,
                 float* Cout, int M, int K, int N) {
  long wn = (long)K * N;
  permute_w_bf16<<<CDIV(wn, 256), 256, 0, st>>>(W, Wbf, K, N);
  dim3 grid(CDIV(M, 256), N / 64);
  gemm_wmma_bf16<<<grid, 256, 0, st>>>(A, Wbf, Cout, M, K, N);
}

static void gcn_layer(hipStream_t st, Bufs& b, const float* X, int n, int Cin, int Cout,
                      const float* W, const float* bias, float* Xout) {
  gemm(st, X, W, b.WBF, b.HB, n, Cin, Cout);
  fill_f32<<<CDIV(n, 256), 256, 0, st>>>(b.deg, 1.f, n);                 // self-loop
  scatter_deg<<<CDIV(NEDGE, 256), 256, 0, st>>>(b.edst, b.emask, b.deg, NEDGE);
  rsqrt_k<<<CDIV(n, 256), 256, 0, st>>>(b.deg, b.dis, n);
  long nc = (long)n * Cout;
  fill_f32<<<CDIV(nc, 256), 256, 0, st>>>(b.AG, 0.f, nc);
  long tE = (long)NEDGE * (Cout / 4);
  gcn_scatter<<<CDIV(tE, 256), 256, 0, st>>>((const float4*)b.HB, b.esrc, b.edst,
                                             b.emask, b.dis, b.AG, NEDGE, Cout / 4);
  gcn_finalize<<<CDIV(nc, 256), 256, 0, st>>>(b.AG, b.HB, b.dis, bias, Xout, n, Cout);
}

static void rgcn_layer(hipStream_t st, Bufs& b, const float* X, int n, int Cin, int Cout,
                       const float* Wr, const float* root, const float* bias,
                       const int* etype, float* Xout) {
  gemm(st, X, root, b.WBF, b.AG, n, Cin, Cout);                          // x @ Wroot
  for (int r = 0; r < 2; ++r) {
    gemm(st, X, Wr + (size_t)r * Cin * Cout, b.WBF, b.HB, n, Cin, Cout);
    long nc = (long)n * Cout;
    fill_f32<<<CDIV(nc, 256), 256, 0, st>>>(b.SB, 0.f, nc);
    fill_f32<<<CDIV(n, 256), 256, 0, st>>>(b.cnt, 0.f, n);
    long tE = (long)NEDGE * (Cout / 4);
    rgcn_scatter<<<CDIV(tE, 256), 256, 0, st>>>((const float4*)b.HB, b.esrc, b.edst,
                                                b.emask, etype, r, b.SB, NEDGE, Cout / 4);
    rgcn_cnt<<<CDIV(NEDGE, 256), 256, 0, st>>>(b.edst, b.emask, etype, r, b.cnt, NEDGE);
    rgcn_combine<<<CDIV(nc, 256), 256, 0, st>>>(b.AG, b.SB, b.cnt, n, Cout);
  }
  long nc = (long)n * Cout;
  rgcn_finalize<<<CDIV(nc, 256), 256, 0, st>>>(b.AG, bias, Xout, n, Cout);
}

static void sag_pool(hipStream_t st, Bufs& b, const float* Xin, float* Xout,
                     int n, int ng, int k, int Cf,
                     const float* relw, const float* relb, const float* rootw) {
  gemv_k<<<CDIV(n, 256), 256, 0, st>>>(Xin, relw, nullptr, b.tn, n, Cf);
  gemv_k<<<CDIV(n, 256), 256, 0, st>>>(Xin, rootw, nullptr, b.tn2, n, Cf);
  fill_f32<<<CDIV(n, 256), 256, 0, st>>>(b.sagg, 0.f, n);
  scatter_scalar<<<CDIV(NEDGE, 256), 256, 0, st>>>(b.tn, b.esrc, b.edst, b.emask, b.sagg, NEDGE);
  score_k<<<CDIV(n, 256), 256, 0, st>>>(b.sagg, b.tn2, relb, b.score, n);
  topk_sort<<<BATCH, 1024, 0, st>>>(b.score, ng, k, b.perm, b.newidx);
  long tg = (long)BATCH * k * Cf;
  pool_gather<<<CDIV(tg, 256), 256, 0, st>>>(Xin, b.score, b.perm, Xout, BATCH * k, Cf);
  remap_edges<<<CDIV(NEDGE, 256), 256, 0, st>>>(b.esrc, b.edst, b.emask, b.newidx, NEDGE);
}

static void readout(hipStream_t st, Bufs& b, const float* X, const float* gw,
                    const float* gb, int k, int colOff) {
  gemv_k<<<CDIV(BATCH * k, 256), 256, 0, st>>>(X, gw, gb, b.gate, BATCH * k, 256);
  softmax_pg<<<BATCH, 256, 0, st>>>(b.gate, k);
  att_wsum<<<dim3(CDIV(256, 64), BATCH), 64, 0, st>>>(X, b.gate, b.R, k, 256, colOff);
}

struct GcnP {
  const float *W1, *b1, *W2, *b2, *W3, *b3;
  const float *p1rw, *p1rb, *p1qw, *p2rw, *p2rb, *p2qw, *gw, *gb;
};

static void run_gcn_view(hipStream_t st, Bufs& b, const float* x, const int* ei,
                         const GcnP& p, int colOff) {
  init_edges<<<CDIV(NEDGE, 256), 256, 0, st>>>(ei, b.esrc, b.edst, b.emask, NEDGE);
  gcn_layer(st, b, x, N1, DIM, 1024, p.W1, p.b1, b.XB);
  sag_pool(st, b, b.XB, b.XA, N1, NPG, KP1, 1024, p.p1rw, p.p1rb, p.p1qw);
  gcn_layer(st, b, b.XA, N2, 1024, 512, p.W2, p.b2, b.XB);
  sag_pool(st, b, b.XB, b.XA, N2, KP1, KP2, 512, p.p2rw, p.p2rb, p.p2qw);
  gcn_layer(st, b, b.XA, N3, 512, 256, p.W3, p.b3, b.XB);
  readout(st, b, b.XB, p.gw, p.gb, KP2, colOff);
}

extern "C" void kernel_launch(void* const* d_in, const int* in_sizes, int n_in,
                              void* d_out, int out_size, void* d_ws, size_t ws_size,
                              hipStream_t stream) {
  (void)in_sizes; (void)n_in; (void)out_size; (void)ws_size;
  const float* ast_x    = (const float*)d_in[0];
  const int*   ast_ei   = (const int*)d_in[1];
  const float* cfg_x    = (const float*)d_in[2];
  const int*   cfg_ei   = (const int*)d_in[3];
  const float* pdg_x    = (const float*)d_in[4];
  const int*   pdg_ei   = (const int*)d_in[5];
  const int*   pdg_et   = (const int*)d_in[6];
  const float* code_emb = (const float*)d_in[7];

  auto gcnP = [&](int base) {
    GcnP p;
    p.W1 = (const float*)d_in[base + 0];  p.b1 = (const float*)d_in[base + 1];
    p.W2 = (const float*)d_in[base + 2];  p.b2 = (const float*)d_in[base + 3];
    p.W3 = (const float*)d_in[base + 4];  p.b3 = (const float*)d_in[base + 5];
    p.p1rw = (const float*)d_in[base + 6]; p.p1rb = (const float*)d_in[base + 7];
    p.p1qw = (const float*)d_in[base + 8];
    p.p2rw = (const float*)d_in[base + 9]; p.p2rb = (const float*)d_in[base + 10];
    p.p2qw = (const float*)d_in[base + 11];
    p.gw = (const float*)d_in[base + 12]; p.gb = (const float*)d_in[base + 13];
    return p;
  };
  GcnP pa = gcnP(8), pc = gcnP(22);

  const float* pdg_W1 = (const float*)d_in[36];
  const float* pdg_r1 = (const float*)d_in[37];
  const float* pdg_b1 = (const float*)d_in[38];
  const float* pdg_W2 = (const float*)d_in[39];
  const float* pdg_r2 = (const float*)d_in[40];
  const float* pdg_b2 = (const float*)d_in[41];
  const float* pdg_W3 = (const float*)d_in[42];
  const float* pdg_r3 = (const float*)d_in[43];
  const float* pdg_b3 = (const float*)d_in[44];
  const float* pdg_p1rw = (const float*)d_in[45];
  const float* pdg_p1rb = (const float*)d_in[46];
  const float* pdg_p1qw = (const float*)d_in[47];
  const float* pdg_p2rw = (const float*)d_in[48];
  const float* pdg_p2rb = (const float*)d_in[49];
  const float* pdg_p2qw = (const float*)d_in[50];
  const float* pdg_gw   = (const float*)d_in[51];
  const float* pdg_gb   = (const float*)d_in[52];
  const float* llm_w = (const float*)d_in[53];
  const float* llm_b = (const float*)d_in[54];
  const float* mlp_w1 = (const float*)d_in[55]; const float* mlp_b1 = (const float*)d_in[56];
  const float* bn1_g  = (const float*)d_in[57]; const float* bn1_b  = (const float*)d_in[58];
  const float* mlp_w2 = (const float*)d_in[59]; const float* mlp_b2 = (const float*)d_in[60];
  const float* bn2_g  = (const float*)d_in[61]; const float* bn2_b  = (const float*)d_in[62];
  const float* mlp_w3 = (const float*)d_in[63]; const float* mlp_b3 = (const float*)d_in[64];
  const float* bn3_g  = (const float*)d_in[65]; const float* bn3_b  = (const float*)d_in[66];
  const float* cls_w  = (const float*)d_in[67]; const float* cls_b  = (const float*)d_in[68];

  // ---- bump allocator over d_ws ----
  char* ws = (char*)d_ws;
  size_t off = 0;
  auto alloc = [&](size_t bytes) -> void* {
    void* p = ws + off;
    off = (off + bytes + 255) & ~(size_t)255;
    return p;
  };
  const size_t BIG = (size_t)N1 * 1024 * sizeof(float);
  Bufs b;
  b.XA = (float*)alloc(BIG);
  b.XB = (float*)alloc(BIG);
  b.HB = (float*)alloc(BIG);
  b.AG = (float*)alloc(BIG);
  b.SB = (float*)alloc(BIG);
  b.WBF = (__bf16*)alloc((size_t)1024 * 1024 * sizeof(__bf16));
  b.deg   = (float*)alloc(N1 * sizeof(float));
  b.dis   = (float*)alloc(N1 * sizeof(float));
  b.cnt   = (float*)alloc(N1 * sizeof(float));
  b.tn    = (float*)alloc(N1 * sizeof(float));
  b.tn2   = (float*)alloc(N1 * sizeof(float));
  b.sagg  = (float*)alloc(N1 * sizeof(float));
  b.score = (float*)alloc(N1 * sizeof(float));
  b.gate  = (float*)alloc(N1 * sizeof(float));
  b.perm   = (int*)alloc((size_t)BATCH * KP1 * sizeof(int));
  b.newidx = (int*)alloc(N1 * sizeof(int));
  b.esrc   = (int*)alloc(NEDGE * sizeof(int));
  b.edst   = (int*)alloc(NEDGE * sizeof(int));
  b.emask  = (float*)alloc(NEDGE * sizeof(float));
  b.R      = (float*)alloc((size_t)BATCH * 1024 * sizeof(float));
  float* Z1 = (float*)alloc((size_t)BATCH * 1024 * sizeof(float));
  float* Z2 = (float*)alloc((size_t)BATCH * 512 * sizeof(float));
  float* Z3 = (float*)alloc((size_t)BATCH * 256 * sizeof(float));

  // ---- three graph views ----
  run_gcn_view(stream, b, ast_x, ast_ei, pa, 0);
  run_gcn_view(stream, b, cfg_x, cfg_ei, pc, 256);

  init_edges<<<CDIV(NEDGE, 256), 256, 0, stream>>>(pdg_ei, b.esrc, b.edst, b.emask, NEDGE);
  rgcn_layer(stream, b, pdg_x, N1, DIM, 1024, pdg_W1, pdg_r1, pdg_b1, pdg_et, b.XB);
  sag_pool(stream, b, b.XB, b.XA, N1, NPG, KP1, 1024, pdg_p1rw, pdg_p1rb, pdg_p1qw);
  rgcn_layer(stream, b, b.XA, N2, 1024, 512, pdg_W2, pdg_r2, pdg_b2, pdg_et, b.XB);
  sag_pool(stream, b, b.XB, b.XA, N2, KP1, KP2, 512, pdg_p2rw, pdg_p2rb, pdg_p2qw);
  rgcn_layer(stream, b, b.XA, N3, 512, 256, pdg_W3, pdg_r3, pdg_b3, pdg_et, b.XB);
  readout(stream, b, b.XB, pdg_gw, pdg_gb, KP2, 512);

  // ---- LLM branch into R[:, 768:1024] ----
  small_gemm<<<CDIV(BATCH * 256, 256), 256, 0, stream>>>(code_emb, llm_w, llm_b, b.R,
                                                         BATCH, DIM, 256, 1024, 768);
  // ---- MLP head ----
  small_gemm<<<CDIV(BATCH * 1024, 256), 256, 0, stream>>>(b.R, mlp_w1, mlp_b1, Z1,
                                                          BATCH, 1024, 1024, 1024, 0);
  bn_leaky<<<CDIV(1024, 256), 256, 0, stream>>>(Z1, bn1_g, bn1_b, BATCH, 1024);
  small_gemm<<<CDIV(BATCH * 512, 256), 256, 0, stream>>>(Z1, mlp_w2, mlp_b2, Z2,
                                                         BATCH, 1024, 512, 512, 0);
  bn_leaky<<<CDIV(512, 256), 256, 0, stream>>>(Z2, bn2_g, bn2_b, BATCH, 512);
  small_gemm<<<CDIV(BATCH * 256, 256), 256, 0, stream>>>(Z2, mlp_w3, mlp_b3, Z3,
                                                         BATCH, 512, 256, 256, 0);
  bn_leaky<<<CDIV(256, 256), 256, 0, stream>>>(Z3, bn3_g, bn3_b, BATCH, 256);
  small_gemm<<<CDIV(BATCH * 2, 256), 256, 0, stream>>>(Z3, cls_w, cls_b, (float*)d_out,
                                                       BATCH, 256, 2, 2, 0);
}
